// HybridMambaNet_14903536517538
// MI455X (gfx1250) — matrix operand order repaired and checked by reading, not compiled, not run on hardware
//
#include <hip/hip_runtime.h>
#include <hip/hip_bf16.h>
#include <stdint.h>

// ---------------- model dims ----------------
#define LB     336      // LOOKBACK
#define NF     164      // N_FEATURES
#define HZ     96       // HORIZON
#define DM     64       // D_MODEL
#define DST    16       // D_STATE
#define DI     128      // D_INNER
#define DTR    5        // DT_RANK
#define BATCH  512
#define NROWS  (BATCH * LB)   // 172032, multiple of 16; tiles never cross batch (336 % 16 == 0)
#define XDBL_W 40             // 37 (dt|B|C) padded to 40 floats -> 160B rows, 16B aligned

typedef __bf16 bf16_t;
typedef __attribute__((ext_vector_type(16))) __bf16 v16bf;
typedef __attribute__((ext_vector_type(8)))  float  v8f;

// v_wmma_f32_16x16x32_bf16: D(16x16 f32) = A(16x32 bf16) * B(32x16 bf16) + C
__device__ __forceinline__ v8f wmma_bf16(v16bf a, v16bf b, v8f c) {
  return __builtin_amdgcn_wmma_f32_16x16x32_bf16(false, a, false, b, (short)0, c, false, false);
}

// A-fragment K index for element e (0..15), lane half = lane>>4 (ISA 7.12.2, 16-bit A 16x32)
__device__ __forceinline__ int amap(int e, int half) {
  return ((e >> 3) << 4) + (half << 3) + (e & 7);
}
// B-fragment K index: k = ((e>>3)<<4) + half*8 + (e&7), same order as amap (16-bit B 32x16)

__device__ __forceinline__ float silu_f(float x) { return x / (1.f + __expf(-x)); }
__device__ __forceinline__ float softplus_f(float x) {
  return (x > 20.f) ? x : __logf(1.f + __expf(x));
}

// Stage a (Kreal x Nreal) f32 weight matrix into LDS as bf16 WMMA B-fragments,
// pre-swizzled so a lane reads its whole v16bf operand contiguously (2x ds_load_b128).
// Layout: Bsw[(kc*NT + nt)*32 + lane][16].  Branchless zero padding via clamp+select.
__device__ __forceinline__ void stage_B_frags(
    const float* __restrict__ W, int Kreal, int Nreal, int stride,
    int KC, int NT, bf16_t* __restrict__ Bsw, int nthr)
{
  const int total = KC * NT * 32;
  for (int fl = threadIdx.x; fl < total; fl += nthr) {
    const int f    = fl >> 5;
    const int lane = fl & 31;
    const int kc   = f / NT;
    const int nt   = f - kc * NT;
    const int half = lane >> 4;
    const int m    = lane & 15;
    const int n    = nt * 16 + m;
    const int ncl  = (n < Nreal) ? n : 0;
    bf16_t* dst = Bsw + fl * 16;
    #pragma unroll
    for (int e = 0; e < 16; ++e) {
      const int k   = kc * 32 + ((e >> 3) << 4) + half * 8 + (e & 7);
      const int kcl = (k < Kreal) ? k : 0;
      const float w = W[kcl * stride + ncl];                 // unconditional load
      dst[e] = (bf16_t)(((k < Kreal) && (n < Nreal)) ? w : 0.f);  // v_cndmask, no branch
    }
  }
}

__device__ __forceinline__ v16bf load_B_frag(const bf16_t* Bsw, int f, int lane) {
  return *((const v16bf*)(Bsw + (size_t)(f * 32 + lane) * 16));
}

// ============================================================================
// Kernel 0: fold embed into in_W (rank-1 collapse) and precompute A = -exp(A_log)
// ============================================================================
__global__ __launch_bounds__(256) void k_prep(
    const float* __restrict__ embed_W, const float* __restrict__ embed_b,
    const float* __restrict__ in_W,    const float* __restrict__ A_log,
    float* __restrict__ wxz, float* __restrict__ bxz, float* __restrict__ Aneg)
{
  int j = threadIdx.x;            // 0..255 -> column of in_W
  float sw = 0.f, sb = 0.f;
  for (int d = 0; d < DM; ++d) {
    float w = in_W[d * (2 * DI) + j];
    sw += embed_W[d] * w;
    sb += embed_b[d] * w;
  }
  wxz[j] = sw;
  bxz[j] = sb;
  for (int idx = j; idx < DI * DST; idx += 256)
    Aneg[idx] = -__expf(A_log[idx]);
}

// ============================================================================
// Kernel 1: x_dbl = silu(depthwise_causal_conv(x)) @ xproj_W   via WMMA bf16
//   one wave per 16-row (b,l) tile; K = 128 = 4x32 ; N = 37 padded to 48 = 3x16
//   - B (xproj_W) staged ONCE per block in LDS, bf16, fragment-swizzled
//   - per-channel conv constants staged in LDS; 4 x_raw taps hoisted per lane
//   - A-fragment elements computed in registers (conv + silu)
// ============================================================================
__global__ __launch_bounds__(256) void k_conv_xdbl(
    const float* __restrict__ x_raw,  const float* __restrict__ conv_W,
    const float* __restrict__ conv_b, const float* __restrict__ xproj_W,
    const float* __restrict__ wxz,    const float* __restrict__ bxz,
    float* __restrict__ xdbl)
{
  __shared__ __align__(32) bf16_t Bsw[12 * 32 * 16];   // 12 KB: 4 kc x 3 nt frags
  __shared__ float chan[DI][8];                        // cw0..3, wx, bx, cb

  // stage channel constants (one row per inner channel)
  for (int dch = threadIdx.x; dch < DI; dch += 256) {
    chan[dch][0] = conv_W[dch * 4 + 0];
    chan[dch][1] = conv_W[dch * 4 + 1];
    chan[dch][2] = conv_W[dch * 4 + 2];
    chan[dch][3] = conv_W[dch * 4 + 3];
    chan[dch][4] = wxz[dch];
    chan[dch][5] = bxz[dch];
    chan[dch][6] = conv_b[dch];
    chan[dch][7] = 0.f;
  }
  stage_B_frags(xproj_W, DI, 37, 37, 4, 3, Bsw, 256);
  __syncthreads();

  const int wave = threadIdx.x >> 5;
  const int lane = threadIdx.x & 31;
  const int tile = blockIdx.x * 8 + wave;
  const int row0 = tile * 16;
  const int half = lane >> 4;
  const int m    = lane & 15;
  const int row  = row0 + m;            // this lane's A row
  const int b    = row / LB;
  const int l    = row % LB;
  const float* xr = x_raw + b * LB;

  // hoist the 4 causal taps for this lane's row (branchless clamp + mask)
  float tap[4], tmk[4];
  #pragma unroll
  for (int kk = 0; kk < 4; ++kk) {
    const int t  = l - 3 + kk;
    const int tc = (t >= 0) ? t : 0;
    const float v = xr[tc];
    tmk[kk] = (t >= 0) ? 1.f : 0.f;
    tap[kk] = v * tmk[kk];
  }

  v8f acc[3];
  #pragma unroll
  for (int nt = 0; nt < 3; ++nt)
    #pragma unroll
    for (int i = 0; i < 8; ++i) acc[nt][i] = 0.f;

  #pragma unroll
  for (int kc = 0; kc < 4; ++kc) {
    // build A fragment: xc(row, d) = silu(wx*conv(x_raw) + bx*sum(valid cw) + cb)
    v16bf afrag;
    #pragma unroll
    for (int e = 0; e < 16; ++e) {
      const int d = kc * 32 + amap(e, half);
      const float* ch = chan[d];
      const float dotc = ch[0]*tap[0] + ch[1]*tap[1] + ch[2]*tap[2] + ch[3]*tap[3];
      const float cwv  = ch[0]*tmk[0] + ch[1]*tmk[1] + ch[2]*tmk[2] + ch[3]*tmk[3];
      const float s = ch[4] * dotc + ch[5] * cwv + ch[6];
      afrag[e] = (bf16_t)silu_f(s);
    }
    #pragma unroll
    for (int nt = 0; nt < 3; ++nt) {
      v16bf bfrag = load_B_frag(Bsw, kc * 3 + nt, lane);
      acc[nt] = wmma_bf16(afrag, bfrag, acc[nt]);
    }
  }
  // store C frags: lane covers column n, rows half*8 + i   (padded row stride 40)
  #pragma unroll
  for (int nt = 0; nt < 3; ++nt) {
    const int n = nt * 16 + m;
    if (n < XDBL_W) {
      #pragma unroll
      for (int i = 0; i < 8; ++i) {
        const int r = row0 + half * 8 + i;
        xdbl[(size_t)r * XDBL_W + n] = acc[nt][i];
      }
    }
  }
}

// ============================================================================
// Kernel 2: selective-state scan, one block per batch element, thread = channel d.
//   Per step: 37 floats of x_dbl streamed HBM->LDS with async double buffering
//   (global_load_async_to_lds_b128 + s_wait_asynccnt), delta & conv recomputed
//   in-register (cheaper than 88MB of materialized delta).
// ============================================================================
__global__ __launch_bounds__(128) void k_scan(
    const float* __restrict__ x_raw,  const float* __restrict__ xdbl,
    const float* __restrict__ dt_W,   const float* __restrict__ dt_b,
    const float* __restrict__ Aneg,   const float* __restrict__ Dvec,
    const float* __restrict__ conv_W, const float* __restrict__ conv_b,
    const float* __restrict__ wxz,    const float* __restrict__ bxz,
    float* __restrict__ ygate)
{
  __shared__ float xrs[LB + 3];        // zero-padded x_raw row
  __shared__ float buf[2][XDBL_W];     // async double buffer (160B each, 16B aligned)

  const int b = blockIdx.x;
  const int d = threadIdx.x;           // 0..127 channel

  for (int i = threadIdx.x; i < LB; i += 128) xrs[3 + i] = x_raw[b * LB + i];
  if (threadIdx.x < 3) xrs[threadIdx.x] = 0.f;

  // per-channel constants
  float dw[DTR];
  #pragma unroll
  for (int r = 0; r < DTR; ++r) dw[r] = dt_W[r * DI + d];
  const float dbias = dt_b[d];
  float a[DST];
  #pragma unroll
  for (int n = 0; n < DST; ++n) a[n] = Aneg[d * DST + n];
  float cw[4];
  #pragma unroll
  for (int k = 0; k < 4; ++k) cw[k] = conv_W[d * 4 + k];
  const float cb = conv_b[d];
  const float wx = wxz[d], bx = bxz[d];
  const float cwsum = cw[0] + cw[1] + cw[2] + cw[3];

  float h[DST];
  #pragma unroll
  for (int n = 0; n < DST; ++n) h[n] = 0.f;
  float ylast = 0.f, xclast = 0.f;

  const float* src = xdbl + (size_t)b * LB * XDBL_W;
  const uint32_t ldsbase = (uint32_t)(uintptr_t)(&buf[0][0]);

  // prefetch t = 0 : 10 lanes x 16B = 160B row
  if (threadIdx.x < 10) {
    const float* p = src + threadIdx.x * 4;
    asm volatile("global_load_async_to_lds_b128 %0, %1, off"
                 :: "v"(ldsbase + threadIdx.x * 16), "v"(p) : "memory");
  }

  for (int t = 0; t < LB; ++t) {
    const int cur = t & 1;
    if (threadIdx.x < 10) {
      if (t + 1 < LB) {
        const float* p = src + (size_t)(t + 1) * XDBL_W + threadIdx.x * 4;
        const uint32_t dst = ldsbase + (uint32_t)(((t + 1) & 1) * (XDBL_W * 4)) + threadIdx.x * 16;
        asm volatile("global_load_async_to_lds_b128 %0, %1, off" :: "v"(dst), "v"(p) : "memory");
        asm volatile("s_wait_asynccnt 0x1" ::: "memory");   // buf[cur] complete, next in flight
      } else {
        asm volatile("s_wait_asynccnt 0x0" ::: "memory");
      }
    }
    __syncthreads();                                        // buf[cur] visible to all waves

    // delta = softplus(dt . dt_W + dt_b)
    float dtv = dbias;
    #pragma unroll
    for (int r = 0; r < DTR; ++r) dtv += buf[cur][r] * dw[r];
    const float delta = softplus_f(dtv);

    // recompute conv+silu channel value xc at (b, t, d)
    float s;
    if (t >= 3) {
      const float c = cw[0]*xrs[t] + cw[1]*xrs[t+1] + cw[2]*xrs[t+2] + cw[3]*xrs[t+3];
      s = wx * c + bx * cwsum + cb;
    } else {
      s = cb;
      #pragma unroll
      for (int kk = 0; kk < 4; ++kk) {
        const int tt = t - 3 + kk;
        if (tt >= 0) s += cw[kk] * (xrs[3 + tt] * wx + bx);
      }
    }
    const float xc = silu_f(s);
    const float dx = delta * xc;

    float y = 0.f;
    #pragma unroll
    for (int n = 0; n < DST; ++n) {
      const float dA = __expf(delta * a[n]);                // a = -exp(A_log) < 0
      h[n] = dA * h[n] + dx * buf[cur][DTR + n];            // B_t
      y   += h[n] * buf[cur][DTR + DST + n];                // C_t
    }
    ylast = y; xclast = xc;                                 // only final step survives
    __syncthreads();                                        // reads done before buffer reuse
  }

  // y = (y + x*D) * silu(z_last), z = xr_last * wz + bz
  const float yfin = ylast + xclast * Dvec[d];
  const float zl   = xrs[3 + LB - 1] * wxz[DI + d] + bxz[DI + d];
  ygate[b * DI + d] = yfin * silu_f(zl);
}

// ============================================================================
// Kernel 3: fused  hm = ygate@out_W ; t1 = relu(feats@W1+b1) ; hmlp = t1@W2+b2
//           out = [hm|hmlp] @ head_W + head_b      (all WMMA bf16, f32 accum)
//   one wave per 16-row tile; weight matrices staged sequentially through one
//   LDS fragment buffer (shared by the 4 waves); C->A re-layout via bf16 LDS tiles.
// ============================================================================
__global__ __launch_bounds__(128) void k_head(
    const float* __restrict__ ygate,  const float* __restrict__ out_W,
    const float* __restrict__ feats,  const float* __restrict__ mlp_W1,
    const float* __restrict__ mlp_b1, const float* __restrict__ mlp_W2,
    const float* __restrict__ mlp_b2, const float* __restrict__ head_W,
    const float* __restrict__ head_b, float* __restrict__ out)
{
  __shared__ __align__(32) bf16_t Bsw[24 * 32 * 16];   // 24 KB, max(KC*NT) = 24 frags
  __shared__ bf16_t cat_s[4][16][96];                  // [hm(64) | hmlp(32)]  12 KB
  __shared__ bf16_t t1_s[4][16][64];                   // 8 KB

  const int wave = threadIdx.x >> 5;
  const int lane = threadIdx.x & 31;
  const int tile = blockIdx.x * 4 + wave;
  const int row0 = tile * 16;
  const int half = lane >> 4;
  const int m    = lane & 15;
  bf16_t (*cat)[96] = cat_s[wave];
  bf16_t (*t1)[64]  = t1_s[wave];

  v8f acc[6];

  // ---- G1: hm = ygate(16x128) @ out_W(128x64) ----
  stage_B_frags(out_W, DI, DM, DM, 4, 4, Bsw, 128);
  __syncthreads();
  #pragma unroll
  for (int nt = 0; nt < 4; ++nt)
    #pragma unroll
    for (int i = 0; i < 8; ++i) acc[nt][i] = 0.f;
  #pragma unroll
  for (int kc = 0; kc < 4; ++kc) {
    v16bf af;
    #pragma unroll
    for (int e = 0; e < 16; ++e)
      af[e] = (bf16_t)ygate[(row0 + m) * DI + kc * 32 + amap(e, half)];
    #pragma unroll
    for (int nt = 0; nt < 4; ++nt)
      acc[nt] = wmma_bf16(af, load_B_frag(Bsw, kc * 4 + nt, lane), acc[nt]);
  }
  #pragma unroll
  for (int nt = 0; nt < 4; ++nt) {
    const int n = nt * 16 + m;
    #pragma unroll
    for (int i = 0; i < 8; ++i) cat[half * 8 + i][n] = (bf16_t)acc[nt][i];
  }
  __syncthreads();

  // ---- G2: t1 = relu(feats(16x164->192) @ mlp_W1(164x64) + b1) ----
  stage_B_frags(mlp_W1, NF, DM, DM, 6, 4, Bsw, 128);
  __syncthreads();
  #pragma unroll
  for (int nt = 0; nt < 4; ++nt)
    #pragma unroll
    for (int i = 0; i < 8; ++i) acc[nt][i] = 0.f;
  #pragma unroll
  for (int kc = 0; kc < 6; ++kc) {
    v16bf af;
    #pragma unroll
    for (int e = 0; e < 16; ++e) {
      const int k   = kc * 32 + amap(e, half);
      const int kcl = (k < NF) ? k : 0;
      const float v = feats[(row0 + m) * NF + kcl];          // unconditional load
      af[e] = (bf16_t)((k < NF) ? v : 0.f);                  // select, no branch
    }
    #pragma unroll
    for (int nt = 0; nt < 4; ++nt)
      acc[nt] = wmma_bf16(af, load_B_frag(Bsw, kc * 4 + nt, lane), acc[nt]);
  }
  #pragma unroll
  for (int nt = 0; nt < 4; ++nt) {
    const int n = nt * 16 + m;
    #pragma unroll
    for (int i = 0; i < 8; ++i)
      t1[half * 8 + i][n] = (bf16_t)fmaxf(acc[nt][i] + mlp_b1[n], 0.f);
  }
  __syncthreads();

  // ---- G3: hmlp = t1(16x64) @ mlp_W2(64x32) + b2  -> cat cols 64..95 ----
  stage_B_frags(mlp_W2, DM, 32, 32, 2, 2, Bsw, 128);
  __syncthreads();
  #pragma unroll
  for (int nt = 0; nt < 2; ++nt)
    #pragma unroll
    for (int i = 0; i < 8; ++i) acc[nt][i] = 0.f;
  #pragma unroll
  for (int kc = 0; kc < 2; ++kc) {
    v16bf af;
    #pragma unroll
    for (int e = 0; e < 16; ++e)
      af[e] = t1[m][kc * 32 + amap(e, half)];
    #pragma unroll
    for (int nt = 0; nt < 2; ++nt)
      acc[nt] = wmma_bf16(af, load_B_frag(Bsw, kc * 2 + nt, lane), acc[nt]);
  }
  #pragma unroll
  for (int nt = 0; nt < 2; ++nt) {
    const int n = nt * 16 + m;
    #pragma unroll
    for (int i = 0; i < 8; ++i)
      cat[half * 8 + i][64 + n] = (bf16_t)(acc[nt][i] + mlp_b2[n]);
  }
  __syncthreads();

  // ---- G4: out = cat(16x96) @ head_W(96x96) + head_b ----
  stage_B_frags(head_W, HZ, HZ, HZ, 3, 6, Bsw, 128);
  __syncthreads();
  #pragma unroll
  for (int nt = 0; nt < 6; ++nt)
    #pragma unroll
    for (int i = 0; i < 8; ++i) acc[nt][i] = 0.f;
  #pragma unroll
  for (int kc = 0; kc < 3; ++kc) {
    v16bf af;
    #pragma unroll
    for (int e = 0; e < 16; ++e)
      af[e] = cat[m][kc * 32 + amap(e, half)];
    #pragma unroll
    for (int nt = 0; nt < 6; ++nt)
      acc[nt] = wmma_bf16(af, load_B_frag(Bsw, kc * 6 + nt, lane), acc[nt]);
  }
  #pragma unroll
  for (int nt = 0; nt < 6; ++nt) {
    const int n = nt * 16 + m;
    #pragma unroll
    for (int i = 0; i < 8; ++i)
      out[(size_t)(row0 + half * 8 + i) * HZ + n] = acc[nt][i] + head_b[n];
  }
}

// ============================================================================
// workspace layout (floats): wxz[256] | bxz[256] | Aneg[2048] | ygate[65536] |
//                            xdbl[NROWS*40]  (total ~27.8 MB)
// ============================================================================
extern "C" void kernel_launch(void* const* d_in, const int* in_sizes, int n_in,
                              void* d_out, int out_size, void* d_ws, size_t ws_size,
                              hipStream_t stream) {
  const float* x_raw   = (const float*)d_in[0];
  const float* x_feat  = (const float*)d_in[1];
  const float* embed_W = (const float*)d_in[2];
  const float* embed_b = (const float*)d_in[3];
  const float* in_W    = (const float*)d_in[4];
  const float* conv_W  = (const float*)d_in[5];
  const float* conv_b  = (const float*)d_in[6];
  const float* xproj_W = (const float*)d_in[7];
  const float* dt_W    = (const float*)d_in[8];
  const float* dt_b    = (const float*)d_in[9];
  const float* A_log   = (const float*)d_in[10];
  const float* Dvec    = (const float*)d_in[11];
  const float* out_W   = (const float*)d_in[12];
  const float* mlp_W1  = (const float*)d_in[13];
  const float* mlp_b1  = (const float*)d_in[14];
  const float* mlp_W2  = (const float*)d_in[15];
  const float* mlp_b2  = (const float*)d_in[16];
  const float* head_W  = (const float*)d_in[17];
  const float* head_b  = (const float*)d_in[18];

  float* ws    = (float*)d_ws;
  float* wxz   = ws;
  float* bxz   = ws + 256;
  float* Aneg  = ws + 512;
  float* ygate = ws + 2560;
  float* xdbl  = ws + 68096;

  k_prep<<<1, 256, 0, stream>>>(embed_W, embed_b, in_W, A_log, wxz, bxz, Aneg);
  k_conv_xdbl<<<NROWS / 16 / 8, 256, 0, stream>>>(x_raw, conv_W, conv_b, xproj_W,
                                                  wxz, bxz, xdbl);
  k_scan<<<BATCH, 128, 0, stream>>>(x_raw, xdbl, dt_W, dt_b, Aneg, Dvec,
                                    conv_W, conv_b, wxz, bxz, ygate);
  k_head<<<BATCH / 16 / 4, 128, 0, stream>>>(ygate, out_W, x_feat, mlp_W1, mlp_b1,
                                             mlp_W2, mlp_b2, head_W, head_b,
                                             (float*)d_out);
}
